// MultiHeadAttention_21904333209969
// MI455X (gfx1250) — compile-verified, matrix operand order checked by reading, not run on hardware
//
#include <hip/hip_runtime.h>

// ---------------------------------------------------------------------------
// MHA forward for MI455X (gfx1250): f32 inputs -> f16 WMMA (f32 accum) path.
// Compute-bound (~137 GFLOP, ~84 MB fully L2-resident), so everything runs
// through v_wmma_f32_16x16x32_f16 with fp32 accumulation.
// ---------------------------------------------------------------------------

typedef __attribute__((ext_vector_type(16))) _Float16 v16h;
typedef __attribute__((ext_vector_type(8)))  _Float16 v8h;
typedef __attribute__((ext_vector_type(8)))  float    v8f;
typedef int v4i32 __attribute__((vector_size(16)));

typedef unsigned short u16;
struct __align__(8)  U16x4 { u16 x, y, z, w; };
struct __align__(16) U16x8 { u16 v[8]; };

union Frag16 { v16h v; v8h h[2]; };

#define NUM_B  4
#define SEQ    2048
#define DIM    1024
#define HEADS  16
#define DHEAD  64
#define MROWS  (NUM_B * SEQ)   // 8192

// Async global->LDS staging (CDNA5 GLOBAL_LOAD_ASYNC_TO_LDS_B128, ASYNCcnt).
#if defined(__has_builtin)
#if __has_builtin(__builtin_amdgcn_global_load_async_to_lds_b128) && \
    __has_builtin(__builtin_amdgcn_s_wait_asynccnt)
#define USE_ASYNC_LDS 1
#endif
#endif
#ifndef USE_ASYNC_LDS
#define USE_ASYNC_LDS 0
#endif

static __device__ __forceinline__ u16 f2h(float x) {
  union { _Float16 h; u16 u; } c; c.h = (_Float16)x; return c.u;
}

// ---------------------------------------------------------------------------
// fp32 -> fp16 conversion (workspace staging)
// ---------------------------------------------------------------------------
__global__ void cvt_f32_to_f16(const float* __restrict__ src,
                               u16* __restrict__ dst, int n) {
  int i = blockIdx.x * blockDim.x + threadIdx.x;
  if (i < n) dst[i] = f2h(src[i]);
}

// ---------------------------------------------------------------------------
// Tiled WMMA GEMM: C[M,N] = A[M,K] * B[K,N]  (A,B f16 row-major, f32 accum).
// Block = 128 threads (4 waves), 128x64 output tile, 32 rows per wave
// (8 accumulators/lane = 64 VGPRs), K stepped by 32.
// A tile staged async (layout-preserving 16B runs); B staged transposed
// so B-fragments are contiguous ds_load_b128.
// ---------------------------------------------------------------------------
__global__ void __launch_bounds__(128)
gemm_wmma_f16(const u16* __restrict__ A, const u16* __restrict__ Bm,
              int M, int N, int K,
              const float* __restrict__ bias,
              float* __restrict__ outF,
              u16* __restrict__ outH, int splitHeads)
{
  __shared__ u16 lds_a[128 * 32];  // [row][k]   8 KB
  __shared__ u16 lds_bt[64 * 32];  // [n][k]     4 KB (B transposed)

  const int tid   = threadIdx.x;
  const int lane  = tid & 31;
  const int w     = tid >> 5;
  const int half  = lane >> 4;
  const int lanen = lane & 15;
  const int tileN = blockIdx.x * 64;
  const int tileM = blockIdx.y * 128;
  const int wrow  = w * 32;        // this wave's 32-row strip

  v8f acc0[4] = {};  // rows wrow +  0..15
  v8f acc1[4] = {};  // rows wrow + 16..31

  for (int k0 = 0; k0 < K; k0 += 32) {
    // ---- Stage A: 128 rows x 32 k as 16B chunks (4 per thread). ----
#pragma unroll
    for (int c = 0; c < 4; ++c) {
      int chunk = c * 128 + tid;
      int row = chunk >> 2;          // 4 chunks of 8 f16 per 32-k row
      int kk  = (chunk & 3) * 8;
      const u16* gsrc = &A[(size_t)(tileM + row) * K + k0 + kk];
      u16* ldst = &lds_a[row * 32 + kk];
#if USE_ASYNC_LDS
      __builtin_amdgcn_global_load_async_to_lds_b128(
          (__attribute__((address_space(1))) v4i32*)gsrc,
          (__attribute__((address_space(3))) v4i32*)ldst, 0, 0);
#else
      *(U16x8*)ldst = *(const U16x8*)gsrc;
#endif
    }
    // ---- Stage B transposed: [n][k] so fragments are contiguous k-runs. ----
#pragma unroll
    for (int c = 0; c < 4; ++c) {
      int chunk = c * 128 + tid;
      int kk = chunk >> 4;
      int nn = (chunk & 15) * 4;
      U16x4 d = *(const U16x4*)&Bm[(size_t)(k0 + kk) * N + tileN + nn];
      lds_bt[(nn + 0) * 32 + kk] = d.x;
      lds_bt[(nn + 1) * 32 + kk] = d.y;
      lds_bt[(nn + 2) * 32 + kk] = d.z;
      lds_bt[(nn + 3) * 32 + kk] = d.w;
    }
#if USE_ASYNC_LDS
    __builtin_amdgcn_s_wait_asynccnt(0);
#endif
    __syncthreads();

    // A-fragments: lane holds row (lane%16); k-runs [h*8..+7], [16+h*8..+7].
    Frag16 a0, a1;
    a0.h[0] = *(const v8h*)&lds_a[(wrow + lanen) * 32 + half * 8];
    a0.h[1] = *(const v8h*)&lds_a[(wrow + lanen) * 32 + 16 + half * 8];
    a1.h[0] = *(const v8h*)&lds_a[(wrow + 16 + lanen) * 32 + half * 8];
    a1.h[1] = *(const v8h*)&lds_a[(wrow + 16 + lanen) * 32 + 16 + half * 8];

#pragma unroll
    for (int t = 0; t < 4; ++t) {
      Frag16 b;
      b.h[0] = *(const v8h*)&lds_bt[(t * 16 + lanen) * 32 + half * 8];
      b.h[1] = *(const v8h*)&lds_bt[(t * 16 + lanen) * 32 + 16 + half * 8];
      acc0[t] = __builtin_amdgcn_wmma_f32_16x16x32_f16(
          false, a0.v, false, b.v, (short)0, acc0[t], false, false);
      acc1[t] = __builtin_amdgcn_wmma_f32_16x16x32_f16(
          false, a1.v, false, b.v, (short)0, acc1[t], false, false);
    }
    __syncthreads();
  }

  // Epilogue. C layout: lane holds col lanen of rows r + 8*half.
#pragma unroll
  for (int s = 0; s < 2; ++s) {
#pragma unroll
    for (int t = 0; t < 4; ++t) {
#pragma unroll
      for (int r = 0; r < 8; ++r) {
        int gr = tileM + wrow + s * 16 + r + 8 * half;
        int gc = tileN + t * 16 + lanen;
        float v = (s == 0) ? acc0[t][r] : acc1[t][r];
        if (outF) {
          outF[(size_t)gr * N + gc] = v + (bias ? bias[gc] : 0.0f);
        } else if (splitHeads) {
          int b  = gr >> 11;       // / SEQ
          int sr = gr & (SEQ - 1);
          int hd = gc >> 6;        // / DHEAD
          int dc = gc & 63;
          outH[(((size_t)(b * HEADS + hd)) * SEQ + sr) * DHEAD + dc] = f2h(v);
        } else {
          outH[(size_t)gr * N + gc] = f2h(v);
        }
      }
    }
  }
}

// ---------------------------------------------------------------------------
// Flash attention (causal): grid (S/64, B*H), 4 waves/block, 16 q-rows/wave.
// k-loop in 32-column tiles up to the causal frontier of the block.
// Q, K fragments loaded straight from global (dh contiguous -> aligned b128);
// V staged transposed in LDS (shared by the block); P bounced through
// per-wave LDS to convert C-tile layout -> A-fragment layout.
// ---------------------------------------------------------------------------
__global__ void __launch_bounds__(128)
flash_attn_wmma(const u16* __restrict__ qh, const u16* __restrict__ kh,
                const u16* __restrict__ vh, u16* __restrict__ ctx)
{
  __shared__ u16 lds_vt[64 * 32];      // [d][k]
  __shared__ u16 lds_p[4 * 16 * 32];   // per-wave P tile

  const int tid   = threadIdx.x;
  const int lane  = tid & 31;
  const int w     = tid >> 5;
  const int half  = lane >> 4;
  const int lanen = lane & 15;
  const int bh    = blockIdx.y;
  const int qbase = blockIdx.x * 64;
  const int qrow0 = qbase + w * 16;
  const size_t headBase = (size_t)bh * SEQ * DHEAD;

  // Q fragments: 2 chunks over dh=64 (K-dim of Q*K^T).
  Frag16 aq[2];
  {
    const u16* qp = qh + headBase + (size_t)(qrow0 + lanen) * DHEAD;
#pragma unroll
    for (int c = 0; c < 2; ++c) {
      aq[c].h[0] = *(const v8h*)&qp[c * 32 + half * 8];
      aq[c].h[1] = *(const v8h*)&qp[c * 32 + 16 + half * 8];
    }
  }

  v8f o[4] = {};
  float mrow[8], lrow[8];
#pragma unroll
  for (int r = 0; r < 8; ++r) { mrow[r] = -1e30f; lrow[r] = 0.0f; }

  const int nkt = (qbase >> 5) + 2;  // 32-wide key tiles up to causal frontier
  for (int kt = 0; kt < nkt; ++kt) {
    const int k0 = kt * 32;

    // Stage V tile transposed: lds_vt[d][k].
#pragma unroll
    for (int c = 0; c < 4; ++c) {
      int chunk = c * 128 + tid;
      int kk = chunk >> 4;
      int nn = (chunk & 15) * 4;
      U16x4 d = *(const U16x4*)&vh[headBase + (size_t)(k0 + kk) * DHEAD + nn];
      lds_vt[(nn + 0) * 32 + kk] = d.x;
      lds_vt[(nn + 1) * 32 + kk] = d.y;
      lds_vt[(nn + 2) * 32 + kk] = d.z;
      lds_vt[(nn + 3) * 32 + kk] = d.w;
    }
    if (kt + 1 < nkt)
      __builtin_prefetch(&vh[headBase + (size_t)(k0 + 32) * DHEAD], 0, 1);
    __syncthreads();

    // Scores: two 16x16 f32 tiles (key cols k0..k0+15 and k0+16..k0+31).
    v8f s0 = {}, s1 = {};
#pragma unroll
    for (int c = 0; c < 2; ++c) {
      Frag16 b0, b1;
      const u16* kp  = kh + headBase + (size_t)(k0 + lanen) * DHEAD + c * 32;
      const u16* kp1 = kp + 16 * DHEAD;
      b0.h[0] = *(const v8h*)&kp[half * 8];
      b0.h[1] = *(const v8h*)&kp[16 + half * 8];
      b1.h[0] = *(const v8h*)&kp1[half * 8];
      b1.h[1] = *(const v8h*)&kp1[16 + half * 8];
      s0 = __builtin_amdgcn_wmma_f32_16x16x32_f16(
          false, aq[c].v, false, b0.v, (short)0, s0, false, false);
      s1 = __builtin_amdgcn_wmma_f32_16x16x32_f16(
          false, aq[c].v, false, b1.v, (short)0, s1, false, false);
    }

    // Online softmax (per-row running max m and sum l).
    float p0[8], p1[8], alpha[8];
#pragma unroll
    for (int r = 0; r < 8; ++r) {
      int grow = qrow0 + r + 8 * half;
      float v0 = s0[r] * 0.125f;            // 1/sqrt(dh)
      float v1 = s1[r] * 0.125f;
      if (k0 + lanen      > grow) v0 = -1e30f;  // causal mask
      if (k0 + 16 + lanen > grow) v1 = -1e30f;
      float rmax = fmaxf(v0, v1);
#pragma unroll
      for (int off = 1; off < 16; off <<= 1)
        rmax = fmaxf(rmax, __shfl_xor(rmax, off, 32));
      float mnew = fmaxf(mrow[r], rmax);
      alpha[r] = __expf(mrow[r] - mnew);
      p0[r] = __expf(v0 - mnew);
      p1[r] = __expf(v1 - mnew);
      float rsum = p0[r] + p1[r];
#pragma unroll
      for (int off = 1; off < 16; off <<= 1)
        rsum += __shfl_xor(rsum, off, 32);
      lrow[r] = lrow[r] * alpha[r] + rsum;
      mrow[r] = mnew;
    }
#pragma unroll
    for (int t = 0; t < 4; ++t)
#pragma unroll
      for (int r = 0; r < 8; ++r)
        o[t][r] = o[t][r] * alpha[r];

    // P: C-tile layout -> per-wave LDS -> A-fragment layout.
    u16* pw = &lds_p[w * 16 * 32];
#pragma unroll
    for (int r = 0; r < 8; ++r) {
      int row16 = r + 8 * half;
      pw[row16 * 32 + lanen]      = f2h(p0[r]);
      pw[row16 * 32 + 16 + lanen] = f2h(p1[r]);
    }
    asm volatile("s_wait_dscnt 0" ::: "memory");  // same-wave LDS RAW

    Frag16 ap;
    ap.h[0] = *(const v8h*)&pw[lanen * 32 + half * 8];
    ap.h[1] = *(const v8h*)&pw[lanen * 32 + 16 + half * 8];

#pragma unroll
    for (int t = 0; t < 4; ++t) {
      Frag16 b;
      b.h[0] = *(const v8h*)&lds_vt[(t * 16 + lanen) * 32 + half * 8];
      b.h[1] = *(const v8h*)&lds_vt[(t * 16 + lanen) * 32 + 16 + half * 8];
      o[t] = __builtin_amdgcn_wmma_f32_16x16x32_f16(
          false, ap.v, false, b.v, (short)0, o[t], false, false);
    }
    __syncthreads();
  }

  // Epilogue: ctx[b*S + row][head*64 + col] (f16, row-major [M, DIM]).
  const int b  = bh >> 4;   // / HEADS
  const int hd = bh & 15;
#pragma unroll
  for (int t = 0; t < 4; ++t) {
#pragma unroll
    for (int r = 0; r < 8; ++r) {
      int grow = qrow0 + r + 8 * half;
      int gcol = hd * DHEAD + t * 16 + lanen;
      float v = o[t][r] / lrow[r];
      ctx[(size_t)(b * SEQ + grow) * DIM + gcol] = f2h(v);
    }
  }
}

// ---------------------------------------------------------------------------
// Launch: convert -> QKV GEMMs -> flash attention -> output projection.
// ---------------------------------------------------------------------------
extern "C" void kernel_launch(void* const* d_in, const int* in_sizes, int n_in,
                              void* d_out, int out_size, void* d_ws, size_t ws_size,
                              hipStream_t stream)
{
  const float* x  = (const float*)d_in[0];
  const float* Wq = (const float*)d_in[1];
  const float* Wk = (const float*)d_in[2];
  const float* Wv = (const float*)d_in[3];
  const float* Wo = (const float*)d_in[4];
  const float* bo = (const float*)d_in[5];

  const size_t MD = (size_t)MROWS * DIM;  // 8388608
  const size_t DD = (size_t)DIM * DIM;    // 1048576

  u16* ws  = (u16*)d_ws;
  u16* xh  = ws;            // x in f16
  u16* wqh = xh  + MD;
  u16* wkh = wqh + DD;
  u16* wvh = wkh + DD;
  u16* woh = wvh + DD;
  u16* qh  = woh + DD;      // [b,h,s,dh] f16
  u16* kh  = qh  + MD;
  u16* vh  = kh  + MD;
  u16* cth = vh  + MD;      // ctx [M, DIM] f16

  const int nMD = (int)MD, nDD = (int)DD;
  cvt_f32_to_f16<<<dim3((nMD + 255) / 256), 256, 0, stream>>>(x,  xh,  nMD);
  cvt_f32_to_f16<<<dim3((nDD + 255) / 256), 256, 0, stream>>>(Wq, wqh, nDD);
  cvt_f32_to_f16<<<dim3((nDD + 255) / 256), 256, 0, stream>>>(Wk, wkh, nDD);
  cvt_f32_to_f16<<<dim3((nDD + 255) / 256), 256, 0, stream>>>(Wv, wvh, nDD);
  cvt_f32_to_f16<<<dim3((nDD + 255) / 256), 256, 0, stream>>>(Wo, woh, nDD);

  dim3 gGemm(DIM / 64, MROWS / 128);  // (16, 64)
  gemm_wmma_f16<<<gGemm, 128, 0, stream>>>(xh, wqh, MROWS, DIM, DIM,
                                           nullptr, nullptr, qh, 1);
  gemm_wmma_f16<<<gGemm, 128, 0, stream>>>(xh, wkh, MROWS, DIM, DIM,
                                           nullptr, nullptr, kh, 1);
  gemm_wmma_f16<<<gGemm, 128, 0, stream>>>(xh, wvh, MROWS, DIM, DIM,
                                           nullptr, nullptr, vh, 1);

  flash_attn_wmma<<<dim3(SEQ / 64, NUM_B * HEADS), 128, 0, stream>>>(
      qh, kh, vh, cth);

  gemm_wmma_f16<<<gGemm, 128, 0, stream>>>(cth, woh, MROWS, DIM, DIM,
                                           bo, (float*)d_out, nullptr, 0);
}
